// ProGen3LayerWithCrossAttn_3461743641058
// MI455X (gfx1250) — matrix operand order repaired and checked by reading, not compiled, not run on hardware
//
#include <hip/hip_runtime.h>
#include <math.h>

// ---------------- problem constants ----------------
#define Bq   2
#define Sq   1024
#define Hq   1024
#define NHq  16
#define HDq  64
#define SEq  512
#define Eq   1024
#define NEXPq 8
#define FFq  2048
#define Tq   (Bq*Sq)          // 2048 tokens
#define NEGV (-1e9f)

// ---------------- WMMA types ----------------
typedef __attribute__((ext_vector_type(16))) __bf16 v16bf;
typedef __attribute__((ext_vector_type(8)))  float  v8f;

// ---------------- generic bf16-WMMA GEMM ----------------
// C[m,n] = alpha * sum_k A[m,k] * W(n,k)   (+bias +residual | rowScale-accumulate)
// A: fp32 row-major [M,K] (lda), batched via strideA
// W: opB==0 -> fp32 [N,K] row-major (ldw);  opB==1 -> fp32 [K,N] row-major (ldw)
// outMode: 0 plain  (off = bz*strideC + m*ldc + n)
//          1 scatter to [B,NH,rows,HD] (m = b*rows+s, n = h*HD+d); batch must be 1
//          2 write from batched attn out to [B,S,H]: bz=(b*NH+h), m=s, n=d
#define BM 128
#define BK 32
#define LDSP 40   // padded K stride (bank-conflict pad)

// Block: 256 threads = 8 wave32s, 4 waves along M (32 rows each),
// 2 waves along N (BNt/2 cols each => TN 16-wide subtiles per wave).
template<int BNt, int TN>
__global__ __launch_bounds__(256)
void gemm_bf16_wmma(const float* __restrict__ A, int lda, long long strideA,
                    const float* __restrict__ W, int ldw, long long strideW, int opB,
                    float* __restrict__ C, int ldc, long long strideC,
                    const float* __restrict__ bias,
                    const float* __restrict__ residual,
                    const float* __restrict__ rowScale, int rowScaleStride,
                    float alpha, int M, int N, int K,
                    int outMode, int rowsPerB)
{
    constexpr int BREG = BNt * BK / 256;   // W-tile floats per thread (16 or 8)
    constexpr int TPR  = BK / BREG;        // threads per W row (opB==0)

    __shared__ __bf16 As[2][BM][LDSP];
    __shared__ __bf16 Bs[2][BNt][LDSP];

    const int tid = threadIdx.x;
    const int bz  = blockIdx.z;
    const int m0  = blockIdx.y * BM;
    const int n0  = blockIdx.x * BNt;
    const float* Ab = A + (long long)bz * strideA;
    const float* Wb = W + (long long)bz * strideW;

    const int lane = tid & 31;
    const int wave = tid >> 5;
    const int wm   = (wave >> 1) * 32;          // 4 waves along M
    const int wn   = (wave & 1) * (BNt / 2);    // 2 waves along N
    const int half = lane >> 4;
    const int l15  = lane & 15;

    float aReg[16];
    float bReg[BREG];

    auto loadA = [&](int k0) {
        int r = tid >> 1, c = (tid & 1) * 16;
        const float* src = Ab + (long long)(m0 + r) * lda + k0 + c;
        #pragma unroll
        for (int i = 0; i < 16; ++i) aReg[i] = src[i];
    };
    auto storeA = [&](int buf) {
        int r = tid >> 1, c = (tid & 1) * 16;
        #pragma unroll
        for (int i = 0; i < 16; ++i) As[buf][r][c + i] = (__bf16)aReg[i];
    };
    auto loadB = [&](int k0) {
        if (opB == 0) {
            int r = tid / TPR, c = (tid % TPR) * BREG;
            const float* src = Wb + (long long)(n0 + r) * ldw + k0 + c;
            #pragma unroll
            for (int i = 0; i < BREG; ++i) bReg[i] = src[i];
        } else {
            int kr = tid >> 3, cn = (tid & 7) * BREG;
            const float* src = Wb + (long long)(k0 + kr) * ldw + n0 + cn;
            #pragma unroll
            for (int i = 0; i < BREG; ++i) bReg[i] = src[i];
        }
    };
    auto storeB = [&](int buf) {
        if (opB == 0) {
            int r = tid / TPR, c = (tid % TPR) * BREG;
            #pragma unroll
            for (int i = 0; i < BREG; ++i) Bs[buf][r][c + i] = (__bf16)bReg[i];
        } else {
            int kr = tid >> 3, cn = (tid & 7) * BREG;
            #pragma unroll
            for (int i = 0; i < BREG; ++i) Bs[buf][cn + i][kr] = (__bf16)bReg[i];
        }
    };

    v8f acc[2][TN] = {};

    const int nIter = K / BK;
    // prologue: stage tile 0
    loadA(0); loadB(0);
    storeA(0); storeB(0);
    __syncthreads();

    for (int it = 0; it < nIter; ++it) {
        const int cur = it & 1;
        const bool hasNext = (it + 1) < nIter;
        if (hasNext) { loadA((it + 1) * BK); loadB((it + 1) * BK); }

        // --- build fragments per ISA 7.12.2 wave32 layouts (LDS -> b128 loads) ---
        v16bf afrag[2], bfrag[TN];
        #pragma unroll
        for (int t = 0; t < 2; ++t) {
            int arow = wm + t * 16 + l15;
            #pragma unroll
            for (int v = 0; v < 8; ++v) {
                int kk = ((v < 4) ? 0 : 16) + half * 8 + (v & 3) * 2;
                afrag[t][2 * v]     = As[cur][arow][kk];
                afrag[t][2 * v + 1] = As[cur][arow][kk + 1];
            }
        }
        #pragma unroll
        for (int j = 0; j < TN; ++j) {
            int ncol = wn + j * 16 + l15;
            #pragma unroll
            for (int v = 0; v < 8; ++v) {
                int kk = ((v < 4) ? 0 : 16) + half * 8 + (v & 3) * 2;
                bfrag[j][2 * v]     = Bs[cur][ncol][kk];
                bfrag[j][2 * v + 1] = Bs[cur][ncol][kk + 1];
            }
        }
        #pragma unroll
        for (int i = 0; i < 2; ++i)
            #pragma unroll
            for (int j = 0; j < TN; ++j)
                acc[i][j] = __builtin_amdgcn_wmma_f32_16x16x32_bf16(
                    false, afrag[i], false, bfrag[j], (short)0, acc[i][j], false, false);

        if (hasNext) { storeA(1 - cur); storeB(1 - cur); }
        __syncthreads();
    }

    // --- epilogue (C/D layout: lane -> n = l15, VGPR r -> m = r + 8*half) ---
    #pragma unroll
    for (int i = 0; i < 2; ++i) {
        #pragma unroll
        for (int j = 0; j < TN; ++j) {
            #pragma unroll
            for (int r = 0; r < 8; ++r) {
                int m = m0 + wm + i * 16 + r + 8 * half;
                int n = n0 + wn + j * 16 + l15;
                float val = acc[i][j][r] * alpha;
                long long off;
                if (outMode == 0) {
                    off = (long long)bz * strideC + (long long)m * ldc + n;
                } else if (outMode == 1) {
                    int b = m / rowsPerB, s = m % rowsPerB;
                    int h = n / HDq,      d = n % HDq;
                    off = (((long long)(b * NHq + h) * rowsPerB + s) * HDq) + d;
                } else {
                    int b = bz / NHq, h = bz % NHq;
                    off = ((long long)(b * Sq + m)) * Hq + h * HDq + n;
                }
                if (rowScale) {
                    C[off] += rowScale[(long long)m * rowScaleStride] * val;
                } else {
                    if (bias)     val += bias[n];
                    if (residual) val += residual[off];
                    C[off] = val;
                }
            }
        }
    }
}

// ---------------- RMSNorm ----------------
__global__ __launch_bounds__(256)
void rmsnorm_kernel(const float* __restrict__ x, const float* __restrict__ w,
                    float* __restrict__ out)
{
    __shared__ float red[256];
    long long row = blockIdx.x;
    const float* xr = x + row * Hq;
    float s = 0.f;
    for (int j = threadIdx.x; j < Hq; j += 256) { float v = xr[j]; s += v * v; }
    red[threadIdx.x] = s; __syncthreads();
    for (int st = 128; st > 0; st >>= 1) {
        if (threadIdx.x < st) red[threadIdx.x] += red[threadIdx.x + st];
        __syncthreads();
    }
    float scale = rsqrtf(red[0] / (float)Hq + 1e-6f);
    for (int j = threadIdx.x; j < Hq; j += 256) out[row * Hq + j] = xr[j] * scale * w[j];
}

// ---------------- RoPE on [B,NH,S,HD] (one thread per rotation pair) ----------------
__global__ void rope_kernel(float* __restrict__ q, float* __restrict__ k,
                            const int* __restrict__ pos)
{
    int idx = blockIdx.x * 256 + threadIdx.x;         // over B*NH*S*(HD/2)
    const int HALF = HDq / 2;
    if (idx >= Bq * NHq * Sq * HALF) return;
    int d2 = idx % HALF;
    int s  = (idx / HALF) % Sq;
    int bh = idx / (HALF * Sq);
    int b  = bh / NHq;
    float p = (float)pos[b * Sq + s];
    float inv = powf(10000.0f, -(float)d2 / (float)HALF);
    float f = p * inv;
    float c = cosf(f), sn = sinf(f);
    long long i0 = ((long long)bh * Sq + s) * HDq + d2;
    long long i1 = i0 + HALF;
    float qa = q[i0], qb = q[i1];
    q[i0] = qa * c - qb * sn;
    q[i1] = qb * c + qa * sn;
    float ka = k[i0], kb = k[i1];
    k[i0] = ka * c - kb * sn;
    k[i1] = kb * c + ka * sn;
}

// ---------------- row softmax with causal or key-padding mask ----------------
__global__ __launch_bounds__(256)
void softmax_kernel(float* __restrict__ sc, int cols, int rowsQ, int causal,
                    const int* __restrict__ mask)
{
    __shared__ float red[256];
    long long row = blockIdx.x;
    float* p = sc + row * (long long)cols;
    int qi = (int)(row % rowsQ);
    int b  = (int)(row / ((long long)rowsQ * NHq));
    float vals[4];
    int nIt = (cols + 255) / 256;
    float mx = -3.4e38f;
    for (int it = 0; it < nIt; ++it) {
        int j = it * 256 + threadIdx.x;
        float v = NEGV;
        if (j < cols) {
            v = p[j];
            bool ok = causal ? (j <= qi) : (mask ? (mask[b * cols + j] != 0) : true);
            if (!ok) v = NEGV;
        }
        vals[it] = v;
        mx = fmaxf(mx, v);
    }
    red[threadIdx.x] = mx; __syncthreads();
    for (int st = 128; st > 0; st >>= 1) {
        if (threadIdx.x < st) red[threadIdx.x] = fmaxf(red[threadIdx.x], red[threadIdx.x + st]);
        __syncthreads();
    }
    mx = red[0]; __syncthreads();
    float sum = 0.f;
    for (int it = 0; it < nIt; ++it) {
        int j = it * 256 + threadIdx.x;
        if (j < cols) { vals[it] = __expf(vals[it] - mx); sum += vals[it]; }
    }
    red[threadIdx.x] = sum; __syncthreads();
    for (int st = 128; st > 0; st >>= 1) {
        if (threadIdx.x < st) red[threadIdx.x] += red[threadIdx.x + st];
        __syncthreads();
    }
    float inv = 1.0f / red[0];
    for (int it = 0; it < nIt; ++it) {
        int j = it * 256 + threadIdx.x;
        if (j < cols) p[j] = vals[it] * inv;
    }
}

// ---------------- gate logits + softmax + top2 + renormalize ----------------
__global__ void gate_top2_kernel(const float* __restrict__ h, const float* __restrict__ gw,
                                 float* __restrict__ we)
{
    int t = blockIdx.x * 256 + threadIdx.x;
    if (t >= Tq) return;
    const float* hr = h + (long long)t * Hq;
    float lg[NEXPq];
    for (int e = 0; e < NEXPq; ++e) {
        const float* w = gw + e * Hq;
        float s = 0.f;
        for (int j = 0; j < Hq; ++j) s += hr[j] * w[j];
        lg[e] = s;
    }
    float mx = lg[0];
    for (int e = 1; e < NEXPq; ++e) mx = fmaxf(mx, lg[e]);
    float pr[NEXPq], sum = 0.f;
    for (int e = 0; e < NEXPq; ++e) { pr[e] = __expf(lg[e] - mx); sum += pr[e]; }
    for (int e = 0; e < NEXPq; ++e) pr[e] /= sum;
    int i1 = 0;
    for (int e = 1; e < NEXPq; ++e) if (pr[e] > pr[i1]) i1 = e;
    int i2 = -1;
    for (int e = 0; e < NEXPq; ++e) { if (e == i1) continue; if (i2 < 0 || pr[e] > pr[i2]) i2 = e; }
    float den = pr[i1] + pr[i2];
    for (int e = 0; e < NEXPq; ++e)
        we[(long long)t * NEXPq + e] = (e == i1) ? pr[i1] / den : ((e == i2) ? pr[i2] / den : 0.f);
}

// ---------------- SwiGLU combine: a = silu(a) * b ----------------
__global__ void silu_mul_kernel(float* __restrict__ a, const float* __restrict__ b, long long n)
{
    long long i = (long long)blockIdx.x * 256 + threadIdx.x;
    if (i >= n) return;
    float x = a[i];
    a[i] = (x / (1.0f + __expf(-x))) * b[i];
}

// ---------------- copy (residual init) ----------------
__global__ void copy_kernel(const float* __restrict__ src, float* __restrict__ dst, long long n)
{
    long long i = (long long)blockIdx.x * 256 + threadIdx.x;
    if (i < n) dst[i] = src[i];
}

// ================== host side ==================
static inline dim3 gemm_grid(int M, int N, int bn, int batches) {
    return dim3((unsigned)(N / bn), (unsigned)(M / BM), (unsigned)batches);
}

extern "C" void kernel_launch(void* const* d_in, const int* in_sizes, int n_in,
                              void* d_out, int out_size, void* d_ws, size_t ws_size,
                              hipStream_t stream)
{
    (void)in_sizes; (void)n_in; (void)out_size; (void)ws_size;
    const float* x_in   = (const float*)d_in[0];   // [B,S,H]
    const int*   posids = (const int*)  d_in[1];   // [B,S]
    const float* enc    = (const float*)d_in[2];   // [B,SE,E]
    const int*   encmsk = (const int*)  d_in[3];   // [B,SE]
    const float* in_ln  = (const float*)d_in[4];
    const float* wq     = (const float*)d_in[5];
    const float* wk     = (const float*)d_in[6];
    const float* wv     = (const float*)d_in[7];
    const float* wo     = (const float*)d_in[8];
    const float* cnorm  = (const float*)d_in[9];
    const float* cq_w   = (const float*)d_in[10];
    const float* cq_b   = (const float*)d_in[11];
    const float* ck_w   = (const float*)d_in[12];
    const float* ck_b   = (const float*)d_in[13];
    const float* cv_w   = (const float*)d_in[14];
    const float* cv_b   = (const float*)d_in[15];
    const float* co_w   = (const float*)d_in[16];
    const float* co_b   = (const float*)d_in[17];
    const float* postln = (const float*)d_in[18];
    const float* gatew  = (const float*)d_in[19];
    const float* w1     = (const float*)d_in[20];  // [NEXP,FF,H]
    const float* w2     = (const float*)d_in[21];  // [NEXP,H,FF]
    const float* w3     = (const float*)d_in[22];  // [NEXP,FF,H]
    float* out = (float*)d_out;

    // ---- workspace layout (floats) ----
    float* ws = (float*)d_ws;
    size_t off = 0;
    float* h_ln   = ws + off; off += (size_t)Tq * Hq;               // normalized activations
    float* qb     = ws + off; off += (size_t)Bq * NHq * Sq * HDq;   // [B,NH,S,HD]
    float* kb     = ws + off; off += (size_t)Bq * NHq * Sq * HDq;
    float* vb     = ws + off; off += (size_t)Bq * NHq * Sq * HDq;
    float* scores = ws + off; off += (size_t)Bq * NHq * Sq * Sq;    // reused for cross
    float* o_flat = ws + off; off += (size_t)Tq * Hq;
    float* x1     = ws + off; off += (size_t)Tq * Hq;
    float* x2     = ws + off; off += (size_t)Tq * Hq;
    float* ckb    = ws + off; off += (size_t)Bq * NHq * SEq * HDq;
    float* cvb    = ws + off; off += (size_t)Bq * NHq * SEq * HDq;
    float* we     = ws + off; off += (size_t)Tq * NEXPq;
    float* ffa    = ws + off; off += (size_t)Tq * FFq;
    float* ffb    = ws + off; off += (size_t)Tq * FFq;

    const float iscale = 0.125f;  // 1/sqrt(HD)

    // ======== self-attention ========
    rmsnorm_kernel<<<Tq, 256, 0, stream>>>(x_in, in_ln, h_ln);

    // q/k/v projections, scattered to [B,NH,S,HD]
    gemm_bf16_wmma<128,4><<<gemm_grid(Tq, Hq, 128, 1), 256, 0, stream>>>(
        h_ln, Hq, 0, wq, Hq, 0, 0, qb, 0, 0, nullptr, nullptr, nullptr, 0,
        1.0f, Tq, Hq, Hq, 1, Sq);
    gemm_bf16_wmma<128,4><<<gemm_grid(Tq, Hq, 128, 1), 256, 0, stream>>>(
        h_ln, Hq, 0, wk, Hq, 0, 0, kb, 0, 0, nullptr, nullptr, nullptr, 0,
        1.0f, Tq, Hq, Hq, 1, Sq);
    gemm_bf16_wmma<128,4><<<gemm_grid(Tq, Hq, 128, 1), 256, 0, stream>>>(
        h_ln, Hq, 0, wv, Hq, 0, 0, vb, 0, 0, nullptr, nullptr, nullptr, 0,
        1.0f, Tq, Hq, Hq, 1, Sq);

    {
        int pairs = Bq * NHq * Sq * (HDq / 2);
        rope_kernel<<<(pairs + 255) / 256, 256, 0, stream>>>(qb, kb, posids);
    }

    // scores[b,h] = Q (SxHD) * K^T : batched over B*NH
    gemm_bf16_wmma<128,4><<<gemm_grid(Sq, Sq, 128, Bq * NHq), 256, 0, stream>>>(
        qb, HDq, (long long)Sq * HDq, kb, HDq, (long long)Sq * HDq, 0,
        scores, Sq, (long long)Sq * Sq, nullptr, nullptr, nullptr, 0,
        iscale, Sq, Sq, HDq, 0, Sq);

    softmax_kernel<<<Bq * NHq * Sq, 256, 0, stream>>>(scores, Sq, Sq, 1, nullptr);

    // O[b,h] = P (SxS) * V (SxHD) -> written into [B,S,H]
    gemm_bf16_wmma<64,2><<<gemm_grid(Sq, HDq, 64, Bq * NHq), 256, 0, stream>>>(
        scores, Sq, (long long)Sq * Sq, vb, HDq, (long long)Sq * HDq, 1,
        o_flat, 0, 0, nullptr, nullptr, nullptr, 0,
        1.0f, Sq, HDq, Sq, 2, Sq);

    // x1 = x + O @ wo^T
    gemm_bf16_wmma<128,4><<<gemm_grid(Tq, Hq, 128, 1), 256, 0, stream>>>(
        o_flat, Hq, 0, wo, Hq, 0, 0, x1, Hq, 0, nullptr, x_in, nullptr, 0,
        1.0f, Tq, Hq, Hq, 0, Sq);

    // ======== cross-attention ========
    rmsnorm_kernel<<<Tq, 256, 0, stream>>>(x1, cnorm, h_ln);

    gemm_bf16_wmma<128,4><<<gemm_grid(Tq, Hq, 128, 1), 256, 0, stream>>>(
        h_ln, Hq, 0, cq_w, Hq, 0, 0, qb, 0, 0, cq_b, nullptr, nullptr, 0,
        1.0f, Tq, Hq, Hq, 1, Sq);
    // ck/cv from encoder states [B*SE, E]
    gemm_bf16_wmma<128,4><<<gemm_grid(Bq * SEq, Hq, 128, 1), 256, 0, stream>>>(
        enc, Eq, 0, ck_w, Eq, 0, 0, ckb, 0, 0, ck_b, nullptr, nullptr, 0,
        1.0f, Bq * SEq, Hq, Eq, 1, SEq);
    gemm_bf16_wmma<128,4><<<gemm_grid(Bq * SEq, Hq, 128, 1), 256, 0, stream>>>(
        enc, Eq, 0, cv_w, Eq, 0, 0, cvb, 0, 0, cv_b, nullptr, nullptr, 0,
        1.0f, Bq * SEq, Hq, Eq, 1, SEq);

    // cross scores: (SxHD) x (SExHD)^T, batched
    gemm_bf16_wmma<128,4><<<gemm_grid(Sq, SEq, 128, Bq * NHq), 256, 0, stream>>>(
        qb, HDq, (long long)Sq * HDq, ckb, HDq, (long long)SEq * HDq, 0,
        scores, SEq, (long long)Sq * SEq, nullptr, nullptr, nullptr, 0,
        iscale, Sq, SEq, HDq, 0, Sq);

    softmax_kernel<<<Bq * NHq * Sq, 256, 0, stream>>>(scores, SEq, Sq, 0, encmsk);

    gemm_bf16_wmma<64,2><<<gemm_grid(Sq, HDq, 64, Bq * NHq), 256, 0, stream>>>(
        scores, SEq, (long long)Sq * SEq, cvb, HDq, (long long)SEq * HDq, 1,
        o_flat, 0, 0, nullptr, nullptr, nullptr, 0,
        1.0f, Sq, HDq, SEq, 2, Sq);

    // x2 = x1 + O @ co^T + co_b
    gemm_bf16_wmma<128,4><<<gemm_grid(Tq, Hq, 128, 1), 256, 0, stream>>>(
        o_flat, Hq, 0, co_w, Hq, 0, 0, x2, Hq, 0, co_b, x1, nullptr, 0,
        1.0f, Tq, Hq, Hq, 0, Sq);

    // ======== MoE FFN ========
    rmsnorm_kernel<<<Tq, 256, 0, stream>>>(x2, postln, h_ln);
    gate_top2_kernel<<<(Tq + 255) / 256, 256, 0, stream>>>(h_ln, gatew, we);

    // out starts at residual x2; each expert accumulates we[:,e] * ff
    copy_kernel<<<(Tq * Hq + 255) / 256, 256, 0, stream>>>(x2, out, (long long)Tq * Hq);

    for (int e = 0; e < NEXPq; ++e) {
        const float* w1e = w1 + (size_t)e * FFq * Hq;
        const float* w2e = w2 + (size_t)e * Hq * FFq;
        const float* w3e = w3 + (size_t)e * FFq * Hq;
        gemm_bf16_wmma<128,4><<<gemm_grid(Tq, FFq, 128, 1), 256, 0, stream>>>(
            h_ln, Hq, 0, w1e, Hq, 0, 0, ffa, FFq, 0, nullptr, nullptr, nullptr, 0,
            1.0f, Tq, FFq, Hq, 0, Sq);
        gemm_bf16_wmma<128,4><<<gemm_grid(Tq, FFq, 128, 1), 256, 0, stream>>>(
            h_ln, Hq, 0, w3e, Hq, 0, 0, ffb, FFq, 0, nullptr, nullptr, nullptr, 0,
            1.0f, Tq, FFq, Hq, 0, Sq);
        silu_mul_kernel<<<(int)(((long long)Tq * FFq + 255) / 256), 256, 0, stream>>>(
            ffa, ffb, (long long)Tq * FFq);
        gemm_bf16_wmma<128,4><<<gemm_grid(Tq, Hq, 128, 1), 256, 0, stream>>>(
            ffa, FFq, 0, w2e, FFq, 0, 0, out, Hq, 0, nullptr, nullptr,
            we + e, NEXPq, 1.0f, Tq, Hq, FFq, 0, Sq);
    }
}
